// Head_15745350107387
// MI455X (gfx1250) — matrix operand order repaired and checked by reading, not compiled, not run on hardware
//
#include <hip/hip_runtime.h>

typedef __attribute__((ext_vector_type(16))) _Float16 v16h;
typedef __attribute__((ext_vector_type(8)))  _Float16 v8h;
typedef __attribute__((ext_vector_type(8)))  float    v8f;
typedef __attribute__((ext_vector_type(4)))  float    v4f;

#define NB 4
#define NT 4096
#define ND 1024
#define NH 64

static __device__ __forceinline__ v8f wmma16(v16h a, v16h b, v8f c) {
    // v_wmma_f32_16x16x32_f16: (neg_a, A, neg_b, B, c_mod, C, reuse_a, reuse_b)
    return __builtin_amdgcn_wmma_f32_16x16x32_f16(false, a, false, b, (short)0, c, false, false);
}

// ---------------------------------------------------------------------------
// Kernel 0: transpose + f16-convert the three [1024,64] weight matrices into
// Wt[p][h][k] so projection B-fragments are contiguous 32B loads (L2-resident).
// ---------------------------------------------------------------------------
__global__ __launch_bounds__(256) void wtrans_kernel(
    const float* __restrict__ Wq, const float* __restrict__ Wk,
    const float* __restrict__ Wv, _Float16* __restrict__ WtAll)
{
    const int p = blockIdx.y;
    const float* W = (p == 0) ? Wq : ((p == 1) ? Wk : Wv);
    _Float16* dst = WtAll + (size_t)p * NH * ND;
    int i = blockIdx.x * blockDim.x + threadIdx.x;   // (h,k), k fastest
    if (i < NH * ND) {
        int h = i >> 10;
        int kk = i & (ND - 1);
        dst[i] = (_Float16)W[(size_t)kk * NH + h];   // W is [k][h]
    }
}

// ---------------------------------------------------------------------------
// Kernel 1: projections.  X[B*T,1024] x W[1024,64] -> qh/kh row-major f16,
// vh transposed [b][h][t] f16.  One wave per 16-token tile; 4 N-tiles; K in
// steps of 32 via v_wmma_f32_16x16x32_f16.
// ---------------------------------------------------------------------------
__global__ __launch_bounds__(256) void proj_kernel(
    const float* __restrict__ q, const float* __restrict__ k,
    const float* __restrict__ v, const _Float16* __restrict__ WtAll,
    _Float16* __restrict__ qh, _Float16* __restrict__ kh,
    _Float16* __restrict__ vhT)
{
    const int proj = blockIdx.z;
    const float* X = (proj == 0) ? q : ((proj == 1) ? k : v);
    const _Float16* Wt = WtAll + (size_t)proj * NH * ND;   // [h][k]

    const int wave  = threadIdx.x >> 5;
    const int lane  = threadIdx.x & 31;
    const int laneM = lane & 15;
    const int laneH = lane >> 4;

    const int tile = blockIdx.x * 8 + wave;      // 0..1023
    const int g0   = tile * 16;                  // global token base
    const int b    = g0 / NT;
    const int t0   = g0 - b * NT;

    const float* arow = X + (size_t)(g0 + laneM) * ND;

    v8f acc0 = {}, acc1 = {}, acc2 = {}, acc3 = {};

    for (int k0 = 0; k0 < ND; k0 += 32) {
        // A fragment (contiguous k, same order as B): lane<16 holds K {0..15},
        // lane>=16 holds K {16..31}  (relative to k0)
        const int ka = k0 + laneH * 16;
        v4f x0 = *(const v4f*)(arow + ka);
        v4f x1 = *(const v4f*)(arow + ka + 4);
        v4f x2 = *(const v4f*)(arow + ka + 8);
        v4f x3 = *(const v4f*)(arow + ka + 12);
        v16h a;
#pragma unroll
        for (int i = 0; i < 4; ++i) {
            a[i]      = (_Float16)x0[i];
            a[4 + i]  = (_Float16)x1[i];
            a[8 + i]  = (_Float16)x2[i];
            a[12 + i] = (_Float16)x3[i];
        }
        // B fragments (32x16): lanes 0-15 K=k0..k0+15, lanes 16-31 K=k0+16..31,
        // column = laneM + 16*nt.  Contiguous 32B per lane from Wt[col][.]
        const int kb = k0 + laneH * 16;
        v16h b0 = *(const v16h*)(Wt + (size_t)(laneM +  0) * ND + kb);
        v16h b1 = *(const v16h*)(Wt + (size_t)(laneM + 16) * ND + kb);
        v16h b2 = *(const v16h*)(Wt + (size_t)(laneM + 32) * ND + kb);
        v16h b3 = *(const v16h*)(Wt + (size_t)(laneM + 48) * ND + kb);

        acc0 = wmma16(a, b0, acc0);
        acc1 = wmma16(a, b1, acc1);
        acc2 = wmma16(a, b2, acc2);
        acc3 = wmma16(a, b3, acc3);
    }

    // C layout: VGPR j -> row (j + 8*laneH), col laneM (+16*nt)
    if (proj < 2) {
        _Float16* dst = (proj == 0) ? qh : kh;
#pragma unroll
        for (int j = 0; j < 8; ++j) {
            const int r = j + laneH * 8;
            _Float16* p = dst + (size_t)(g0 + r) * NH + laneM;
            p[0]  = (_Float16)acc0[j];
            p[16] = (_Float16)acc1[j];
            p[32] = (_Float16)acc2[j];
            p[48] = (_Float16)acc3[j];
        }
    } else {
#pragma unroll
        for (int j = 0; j < 8; ++j) {
            const int r = j + laneH * 8;
            const size_t t = (size_t)(t0 + r);
            vhT[((size_t)b * NH + laneM +  0) * NT + t] = (_Float16)acc0[j];
            vhT[((size_t)b * NH + laneM + 16) * NT + t] = (_Float16)acc1[j];
            vhT[((size_t)b * NH + laneM + 32) * NT + t] = (_Float16)acc2[j];
            vhT[((size_t)b * NH + laneM + 48) * NT + t] = (_Float16)acc3[j];
        }
    }
}

// ---------------------------------------------------------------------------
// Kernel 2: causal flash attention.  One wave per 16-query tile, 32 keys per
// iteration.  Scores & P live in registers; P goes C-layout -> A-layout via a
// per-wave private LDS buffer (same-wave LDS is in-order; explicit
// s_wait_dscnt).  Row-sums computed with an extra WMMA against an all-ones
// B fragment (P @ 1 -> row sum in every column), replacing shuffle reductions.
// ---------------------------------------------------------------------------
__global__ __launch_bounds__(256) void attn_kernel(
    const _Float16* __restrict__ qh, const _Float16* __restrict__ kh,
    const _Float16* __restrict__ vhT, float* __restrict__ out)
{
    __shared__ __align__(32) _Float16 pbuf[8][16 * 32];   // 8 KB

    const int wave  = threadIdx.x >> 5;
    const int lane  = threadIdx.x & 31;
    const int laneM = lane & 15;
    const int laneH = lane >> 4;

    const int tile = blockIdx.x * 8 + wave;   // 0..1023
    const int b    = tile >> 8;               // /256
    const int q0   = (tile & 255) * 16;

    // Q A-fragments for head dims 0..31 (a_lo) and 32..63 (a_hi), pre-scaled
    const _Float16* qbase = qh + ((size_t)b * NT + q0 + laneM) * NH;
    v16h a_lo, a_hi;
    {
        const int ka = laneH * 16;
        v8h q0v = *(const v8h*)(qbase + ka);
        v8h q1v = *(const v8h*)(qbase + ka + 8);
        v8h q2v = *(const v8h*)(qbase + 32 + ka);
        v8h q3v = *(const v8h*)(qbase + 32 + ka + 8);
        const _Float16 s = (_Float16)0.125f;   // 1/sqrt(64), exact in f16
#pragma unroll
        for (int i = 0; i < 8; ++i) {
            a_lo[i]     = q0v[i] * s;
            a_lo[8 + i] = q1v[i] * s;
            a_hi[i]     = q2v[i] * s;
            a_hi[8 + i] = q3v[i] * s;
        }
    }

    // all-ones B fragment for row-sum WMMA
    v16h ones;
#pragma unroll
    for (int i = 0; i < 16; ++i) ones[i] = (_Float16)1.0f;

    v8f o0 = {}, o1 = {}, o2 = {}, o3 = {};
    float m[8], l[8];
#pragma unroll
    for (int j = 0; j < 8; ++j) { m[j] = -1e30f; l[j] = 0.0f; }

    const _Float16* krow  = kh  + (size_t)b * NT * NH;
    const _Float16* vbase = vhT + (size_t)b * NH * NT;
    _Float16* pw = &pbuf[wave][0];

    const int kvEnd = q0 + 16;                 // exclusive causal key bound
    for (int kv = 0; kv < kvEnd; kv += 32) {
        // ---- S = Q K^T for key tiles [kv,kv+16) and [kv+16,kv+32) ----
        const int key0 = kv + laneM;
        int key1c = kv + 16 + laneM; key1c = (key1c < NT) ? key1c : (NT - 1);
        const _Float16* k0p = krow + (size_t)key0  * NH + laneH * 16;
        const _Float16* k1p = krow + (size_t)key1c * NH + laneH * 16;
        v16h b0lo = *(const v16h*)(k0p);
        v16h b0hi = *(const v16h*)(k0p + 32);
        v16h b1lo = *(const v16h*)(k1p);
        v16h b1hi = *(const v16h*)(k1p + 32);

        v8f S0 = {}, S1 = {};
        S0 = wmma16(a_lo, b0lo, S0);
        S0 = wmma16(a_hi, b0hi, S0);
        S1 = wmma16(a_lo, b1lo, S1);
        S1 = wmma16(a_hi, b1hi, S1);

        // ---- online softmax per row (row = j + 8*laneH, col = laneM) ----
        float corr[8];
#pragma unroll
        for (int j = 0; j < 8; ++j) {
            const int qrow = q0 + j + laneH * 8;
            float s0 = (kv + laneM      <= qrow) ? S0[j] : -1e30f;
            float s1 = (kv + 16 + laneM <= qrow) ? S1[j] : -1e30f;

            float tmax = fmaxf(s0, s1);
#pragma unroll
            for (int off = 1; off < 16; off <<= 1)
                tmax = fmaxf(tmax, __shfl_xor(tmax, off, 32));

            const float mnew = fmaxf(m[j], tmax);
            corr[j] = __expf(m[j] - mnew);
            m[j] = mnew;
            const float p0 = __expf(s0 - mnew);
            const float p1 = __expf(s1 - mnew);

            o0[j] *= corr[j]; o1[j] *= corr[j]; o2[j] *= corr[j]; o3[j] *= corr[j];

            // stage P in LDS at its (row, col) position (C layout write)
            const int r = j + laneH * 8;
            pw[r * 32 + laneM]      = (_Float16)p0;
            pw[r * 32 + 16 + laneM] = (_Float16)p1;
        }

        // same-wave LDS RAW: wait for our stores, block compiler reordering
        asm volatile("s_wait_dscnt 0" ::: "memory");

        // ---- read P back in A layout (row = laneM, K chunks per laneH) ----
        const int pa0 = laneM * 32 + laneH * 16;
        v8h pl0 = *(const v8h*)(pw + pa0);
        v8h pl1 = *(const v8h*)(pw + pa0 + 8);
        v16h pa;
#pragma unroll
        for (int i = 0; i < 8; ++i) { pa[i] = pl0[i]; pa[8 + i] = pl1[i]; }

        // ---- row sums via WMMA: (P @ ones)[r][c] = sum_k P[r][k] ----
        v8f zero = {};
        v8f rsum = wmma16(pa, ones, zero);
#pragma unroll
        for (int j = 0; j < 8; ++j)
            l[j] = l[j] * corr[j] + rsum[j];

        // ---- O += P V : V B-fragments contiguous along keys in vhT ----
        int keyb = kv + laneH * 16;
        keyb = (keyb <= NT - 16) ? keyb : (NT - 16);   // masked lanes: P==0
        v16h vb0 = *(const v16h*)(vbase + (size_t)(laneM +  0) * NT + keyb);
        v16h vb1 = *(const v16h*)(vbase + (size_t)(laneM + 16) * NT + keyb);
        v16h vb2 = *(const v16h*)(vbase + (size_t)(laneM + 32) * NT + keyb);
        v16h vb3 = *(const v16h*)(vbase + (size_t)(laneM + 48) * NT + keyb);

        o0 = wmma16(pa, vb0, o0);
        o1 = wmma16(pa, vb1, o1);
        o2 = wmma16(pa, vb2, o2);
        o3 = wmma16(pa, vb3, o3);
    }

    // ---- normalize and write fp32 output [b][t][h] ----
    float* orow = out + ((size_t)b * NT + q0) * NH;
#pragma unroll
    for (int j = 0; j < 8; ++j) {
        const int r = j + laneH * 8;
        const float inv = 1.0f / l[j];
        orow[(size_t)r * NH + laneM]      = o0[j] * inv;
        orow[(size_t)r * NH + 16 + laneM] = o1[j] * inv;
        orow[(size_t)r * NH + 32 + laneM] = o2[j] * inv;
        orow[(size_t)r * NH + 48 + laneM] = o3[j] * inv;
    }
}

// ---------------------------------------------------------------------------
extern "C" void kernel_launch(void* const* d_in, const int* in_sizes, int n_in,
                              void* d_out, int out_size, void* d_ws, size_t ws_size,
                              hipStream_t stream) {
    const float* q  = (const float*)d_in[0];
    const float* k  = (const float*)d_in[1];
    const float* v  = (const float*)d_in[2];
    // d_in[3] = mask (int32 causal) -- causality computed analytically
    const float* Wq = (const float*)d_in[4];
    const float* Wk = (const float*)d_in[5];
    const float* Wv = (const float*)d_in[6];

    const size_t BTH = (size_t)NB * NT * NH;   // 1,048,576 halves per tensor
    _Float16* qh    = (_Float16*)d_ws;
    _Float16* kh    = qh  + BTH;
    _Float16* vhT   = kh  + BTH;               // [b][h][t]
    _Float16* WtAll = vhT + BTH;               // 3 x [64][1024] f16

    wtrans_kernel<<<dim3((NH * ND + 255) / 256, 3), 256, 0, stream>>>(Wq, Wk, Wv, WtAll);
    proj_kernel<<<dim3(128, 1, 3), 256, 0, stream>>>(q, k, v, WtAll, qh, kh, vhT);
    attn_kernel<<<dim3(128), 256, 0, stream>>>(qh, kh, vhT, (float*)d_out);
}
